// OrthLorasInjectedLinear_39960375722397
// MI455X (gfx1250) — matrix operand order repaired and checked by reading, not compiled, not run on hardware
//
#include <hip/hip_runtime.h>
#include <hip/hip_bf16.h>

// ---------------------------------------------------------------------------
// OrthLorasInjectedLinear for MI455X (gfx1250, wave32, WMMA + TDM)
//
// out[m,n] = (X @ W0^T)[m,n] + b0[n] + 8 * dot16(h[m, 16l:16l+16], B[l,n,:])
//   where h = X @ A^T ([8192,64]) and l = floor(4*u(m,n)) -- the four
//   keep=0.25 orthogonal-dropout intervals exactly partition [0,1), so
//   exactly one lora fires per element.  coef/keep = (32/16)*4 = 8.
//
// Precision: fp32 -> bf16 hi/lo split, 3x v_wmma_f32_16x16x32_bf16 per tile
// (hi*hi + hi*lo + lo*hi) => ~fp32 accuracy at bf16 matrix rates.
// Data movement: Tensor Data Mover (tensor_load_to_lds) double-buffered on
// TENSORcnt; D# pad feature produces an 80B LDS row pitch (bank spreading).
// ---------------------------------------------------------------------------

typedef __bf16 v16bf __attribute__((ext_vector_type(16)));
typedef __bf16 v8bf  __attribute__((ext_vector_type(8)));
typedef float  v8f   __attribute__((ext_vector_type(8)));
typedef unsigned int v4u __attribute__((ext_vector_type(4)));
typedef int v4i __attribute__((ext_vector_type(4)));
typedef int v8i __attribute__((ext_vector_type(8)));

namespace {
constexpr int MTOT = 8192;   // B*S
constexpr int NTOT = 4096;   // O
constexpr int KTOT = 4096;   // D
constexpr int LR   = 64;     // N_LORAS * RANK
constexpr int KT   = 32;     // K per LDS stage (one bf16 WMMA K)
constexpr int LDT  = 40;     // padded LDS row pitch in bf16 (80 B)
constexpr int NK   = KTOT / KT;           // 128 k-stages
constexpr unsigned PLANE  = 128 * LDT * 2;    // 10240 B per tile plane
constexpr unsigned STAGE  = 4 * PLANE;        // 40960 B per pipeline stage
}

// deterministic stand-in for threefry(key=42) interval draw
static __device__ __forceinline__ unsigned hash_u32(unsigned v) {
  v ^= 0x2545F491u;                 // key-42 fold
  v *= 0x85EBCA6Bu; v ^= v >> 13;
  v *= 0xC2B2AE35u; v ^= v >> 16;
  return v;
}

// ---- WMMA fragment loads (contiguous k order, shared by A and B) --------
// A (16x32 bf16): lane L -> row L&15, K = (L>=16 ? 16 : 0) + c (contiguous).
static __device__ __forceinline__ v16bf load_a_frag(const __bf16* s, int m_off,
                                                    int lane) {
  const __bf16* p = s + (m_off + (lane & 15)) * LDT + ((lane >> 4) << 4);
  v8bf a0 = *(const v8bf*)(p);
  v8bf a1 = *(const v8bf*)(p + 8);
  return __builtin_shufflevector(a0, a1, 0, 1, 2, 3, 4, 5, 6, 7, 8, 9, 10, 11,
                                 12, 13, 14, 15);
}

// B (32x16 bf16): lane L -> column N = L&15, K = (L>=16 ? 16 : 0) + c.
// LDS W-tile is stored [n][k] so this is 32 contiguous bytes.
static __device__ __forceinline__ v16bf load_b_frag(const __bf16* s, int n_off,
                                                    int lane) {
  const __bf16* p = s + (n_off + (lane & 15)) * LDT + ((lane >> 4) << 4);
  v8bf b0 = *(const v8bf*)(p);
  v8bf b1 = *(const v8bf*)(p + 8);
  return __builtin_shufflevector(b0, b1, 0, 1, 2, 3, 4, 5, 6, 7, 8, 9, 10, 11,
                                 12, 13, 14, 15);
}

static __device__ __forceinline__ v8f wmma_bf16(v16bf a, v16bf b, v8f c) {
  return __builtin_amdgcn_wmma_f32_16x16x32_bf16(false, a, false, b, (short)0,
                                                 c, false, false);
}

// ---- Tensor Data Mover: one 128x32-bf16 tile -> LDS (pitch 80 B) ----------
// Tensor viewed as u32 (data_size=4B): 16 dwords per tile row, row stride
// KTOT/2 dwords.  D# pad: after every 16 dwords insert 4 dwords => 80B pitch.
static __device__ __forceinline__ void tdm_load_tile_128x32(
    unsigned lds_byte_off, const void* gtile, unsigned rows_total) {
  const unsigned long long ga = (unsigned long long)(size_t)gtile;
  const unsigned dim0 = (unsigned)(KTOT / 2);  // 2048 dwords per tensor row
  v4u g0 = {1u,                                     // count=1 (valid), user D#
            lds_byte_off,                           // lds_addr
            (unsigned)ga,                           // global_addr[31:0]
            (unsigned)((ga >> 32) & 0x01FFFFFFu) | 0x80000000u};  // type=2
  v8i g1 = {
      (int)((2u << 16) |          // data_size = 4 B
            (1u << 20) |          // pad_enable
            (3u << 22) |          // pad_interval: every 16 dwords
            (3u << 25)),          // pad_amount: 4 dwords
      (int)((dim0 & 0xFFFFu) << 16),                         // dim0 lo16
      (int)((dim0 >> 16) | ((rows_total & 0xFFFFu) << 16)),  // dim0 hi|dim1 lo
      (int)((rows_total >> 16) | (16u << 16)),               // dim1 hi|tile0=16
      128,                        // tile_dim1 = 128 rows (tile_dim2 = 0)
      (int)dim0,                  // tensor_dim0_stride lo32
      0, 0};                      // stride0 hi16, stride1 (unused)
  const v4i z4 = {0, 0, 0, 0};
#if defined(__clang_major__) && (__clang_major__ >= 23)
  const v8i z8 = {0, 0, 0, 0, 0, 0, 0, 0};
  __builtin_amdgcn_tensor_load_to_lds(g0, g1, z4, z4, z8, 0);
#else
  __builtin_amdgcn_tensor_load_to_lds(g0, g1, z4, z4, 0);
#endif
}

// rows x 32 bf16 tile: global (pitch ld) -> LDS (pitch LDT); 2 threads/row.
static __device__ __forceinline__ void copy_tile(__bf16* dst,
                                                 const __bf16* src, int ld,
                                                 int t, int rows) {
  if (t < rows * 2) {
    const int row = t >> 1, half = t & 1;
    const uint4* g = (const uint4*)(src + (size_t)row * ld + half * 16);
    uint4* s = (uint4*)(dst + row * LDT + half * 16);
    s[0] = g[0];
    s[1] = g[1];
  }
}

// ---------------------------------------------------------------------------
// fp32 -> bf16 hi/lo split
// ---------------------------------------------------------------------------
__global__ __launch_bounds__(256) void olil_split_hilo(
    const float* __restrict__ src, __bf16* __restrict__ hi,
    __bf16* __restrict__ lo, long long n) {
  long long i = (long long)blockIdx.x * blockDim.x + threadIdx.x;
  const long long stride = (long long)gridDim.x * blockDim.x;
  for (; i < n; i += stride) {
    const float f = src[i];
    const __bf16 h = (__bf16)f;
    hi[i] = h;
    lo[i] = (__bf16)(f - (float)h);
  }
}

// ---------------------------------------------------------------------------
// h = X @ A^T  -> [MTOT, 64] f32.  Block: 128 rows, 8 waves (16 rows each).
// ---------------------------------------------------------------------------
__global__ __launch_bounds__(256) void olil_lora_h(
    const __bf16* __restrict__ Xhi, const __bf16* __restrict__ Xlo,
    const __bf16* __restrict__ Ahi, const __bf16* __restrict__ Alo,
    float* __restrict__ Hbuf) {
  __shared__ alignas(16) __bf16 sXhi[128 * LDT];
  __shared__ alignas(16) __bf16 sXlo[128 * LDT];
  __shared__ alignas(16) __bf16 sAhi[64 * LDT];
  __shared__ alignas(16) __bf16 sAlo[64 * LDT];

  const int t = threadIdx.x, lane = t & 31, wid = t >> 5;
  const int m0 = blockIdx.x * 128;

  const v8f vzero = {0.f, 0.f, 0.f, 0.f, 0.f, 0.f, 0.f, 0.f};
  v8f acc[4];
#pragma unroll
  for (int ni = 0; ni < 4; ++ni) acc[ni] = vzero;

  const __bf16* gxh = Xhi + (size_t)m0 * KTOT;
  const __bf16* gxl = Xlo + (size_t)m0 * KTOT;

  for (int kt = 0; kt < NK; ++kt) {
    const int k0 = kt * KT;
    copy_tile(sXhi, gxh + k0, KTOT, t, 128);
    copy_tile(sXlo, gxl + k0, KTOT, t, 128);
    copy_tile(sAhi, Ahi + k0, KTOT, t, 64);
    copy_tile(sAlo, Alo + k0, KTOT, t, 64);
    __syncthreads();

    const v16bf ahi = load_a_frag(sXhi, wid * 16, lane);
    const v16bf alo = load_a_frag(sXlo, wid * 16, lane);
#pragma unroll
    for (int ni = 0; ni < 4; ++ni) {
      const v16bf bh = load_b_frag(sAhi, ni * 16, lane);
      const v16bf bl = load_b_frag(sAlo, ni * 16, lane);
      acc[ni] = wmma_bf16(ahi, bh, acc[ni]);
      acc[ni] = wmma_bf16(ahi, bl, acc[ni]);
      acc[ni] = wmma_bf16(alo, bh, acc[ni]);
    }
    __syncthreads();
  }

  const int mbase = wid * 16 + ((lane >> 4) << 3);
#pragma unroll
  for (int ni = 0; ni < 4; ++ni) {
    const int nl = ni * 16 + (lane & 15);
#pragma unroll
    for (int c = 0; c < 8; ++c)
      Hbuf[(size_t)(m0 + mbase + c) * LR + nl] = acc[ni][c];
  }
}

// ---------------------------------------------------------------------------
// Main fused GEMM + orthogonal-lora epilogue.
// Grid (NTOT/128, MTOT/128); 256 threads = 8 waves; wave tile 64x32.
// TDM double-buffer: wave 0 issues 4 tensor_load_to_lds per stage, pipeline
// depth 2, synchronized with s_wait_tensorcnt + workgroup barrier.
// ---------------------------------------------------------------------------
__global__ __launch_bounds__(256) void olil_main_gemm(
    const __bf16* __restrict__ Xhi, const __bf16* __restrict__ Xlo,
    const __bf16* __restrict__ Whi, const __bf16* __restrict__ Wlo,
    const float* __restrict__ b0, const float* __restrict__ Hbuf,
    const float* __restrict__ Blora, float* __restrict__ out) {
  __shared__ alignas(16) unsigned char smem[2 * STAGE];  // 80 KB

  const int t = threadIdx.x, lane = t & 31, wid = t >> 5;
  const int m0 = blockIdx.y * 128, n0 = blockIdx.x * 128;
  const int wm = (wid >> 2) * 64, wn = (wid & 3) * 32;

  const v8f vzero = {0.f, 0.f, 0.f, 0.f, 0.f, 0.f, 0.f, 0.f};
  v8f acc[4][2];
#pragma unroll
  for (int mi = 0; mi < 4; ++mi)
#pragma unroll
    for (int ni = 0; ni < 2; ++ni) acc[mi][ni] = vzero;

  const __bf16* gxh = Xhi + (size_t)m0 * KTOT;
  const __bf16* gxl = Xlo + (size_t)m0 * KTOT;
  const __bf16* gwh = Whi + (size_t)n0 * KTOT;
  const __bf16* gwl = Wlo + (size_t)n0 * KTOT;
  const unsigned ldsBase = (unsigned)(size_t)(void*)smem;  // LDS byte offset

  auto issue_stage = [&](int kt, int stage) {
    const int k0 = kt * KT;
    const unsigned b = ldsBase + (unsigned)stage * STAGE;
    tdm_load_tile_128x32(b + 0 * PLANE, gxh + k0, (unsigned)MTOT);
    tdm_load_tile_128x32(b + 1 * PLANE, gxl + k0, (unsigned)MTOT);
    tdm_load_tile_128x32(b + 2 * PLANE, gwh + k0, (unsigned)NTOT);
    tdm_load_tile_128x32(b + 3 * PLANE, gwl + k0, (unsigned)NTOT);
  };

  if (wid == 0) issue_stage(0, 0);   // prime the pipeline

  for (int kt = 0; kt < NK; ++kt) {
    const int cur = kt & 1;
    if (wid == 0) {
      if (kt + 1 < NK) {
        issue_stage(kt + 1, cur ^ 1);           // stream next stage behind
        __builtin_amdgcn_s_wait_tensorcnt(4);   // current stage (4 ops) done
      } else {
        __builtin_amdgcn_s_wait_tensorcnt(0);   // drain
      }
    }
    __syncthreads();   // stage `cur` visible to all 8 waves

    const __bf16* sXhi = (const __bf16*)(smem + cur * STAGE + 0 * PLANE);
    const __bf16* sXlo = (const __bf16*)(smem + cur * STAGE + 1 * PLANE);
    const __bf16* sWhi = (const __bf16*)(smem + cur * STAGE + 2 * PLANE);
    const __bf16* sWlo = (const __bf16*)(smem + cur * STAGE + 3 * PLANE);

    v16bf bhi[2], blo[2];
#pragma unroll
    for (int ni = 0; ni < 2; ++ni) {
      bhi[ni] = load_b_frag(sWhi, wn + ni * 16, lane);
      blo[ni] = load_b_frag(sWlo, wn + ni * 16, lane);
    }
#pragma unroll
    for (int mi = 0; mi < 4; ++mi) {
      const v16bf ahi = load_a_frag(sXhi, wm + mi * 16, lane);
      const v16bf alo = load_a_frag(sXlo, wm + mi * 16, lane);
#pragma unroll
      for (int ni = 0; ni < 2; ++ni) {
        acc[mi][ni] = wmma_bf16(ahi, bhi[ni], acc[mi][ni]);  // hi*hi
        acc[mi][ni] = wmma_bf16(ahi, blo[ni], acc[mi][ni]);  // hi*lo
        acc[mi][ni] = wmma_bf16(alo, bhi[ni], acc[mi][ni]);  // lo*hi
      }
    }
    __syncthreads();   // stage `cur` free before wave0 re-fills it at kt+2
  }

  // ---- epilogue: reuse LDS for h row-tile (32 KB) + lora-B tile (32 KB) ----
  float* sH = (float*)(smem);            // [128][64]  h rows for this m-tile
  float* sB = (float*)(smem + 32768);    // [128][64]  B[l][n0+n][r] at l*16+r
  {
    const uint4* gh = (const uint4*)(Hbuf + (size_t)m0 * LR);
    uint4* sh4 = (uint4*)sH;
    for (int i = t; i < 128 * LR / 4; i += 256) sh4[i] = gh[i];
    for (int chunk = t; chunk < 512; chunk += 256) {
      const int n = chunk >> 2, l = chunk & 3;
      const uint4* gb =
          (const uint4*)(Blora + ((size_t)l * NTOT + (n0 + n)) * 16);
      uint4* sb = (uint4*)(sB + n * LR + l * 16);
      sb[0] = gb[0]; sb[1] = gb[1]; sb[2] = gb[2]; sb[3] = gb[3];
    }
  }
  __syncthreads();

#pragma unroll
  for (int ni = 0; ni < 2; ++ni) {
    const int nl = wn + ni * 16 + (lane & 15);
    const int n = n0 + nl;
    const float bias = b0[n];
#pragma unroll
    for (int mi = 0; mi < 4; ++mi) {
      const int mbase = wm + mi * 16 + ((lane >> 4) << 3);
      const v8f a = acc[mi][ni];
#pragma unroll
      for (int c = 0; c < 8; ++c) {
        const int ml = mbase + c;
        const unsigned u = hash_u32((unsigned)((m0 + ml) * NTOT + n));
        const int l = u >> 30;  // keep=0.25 x4 partitions [0,1): one lora fires
        const float* hp = sH + ml * LR + l * 16;
        const float* bp = sB + nl * LR + l * 16;
        float y = 0.f;
#pragma unroll
        for (int r = 0; r < 16; ++r) y += hp[r] * bp[r];
        out[(size_t)(m0 + ml) * NTOT + n] = a[c] + bias + 8.0f * y;
      }
    }
  }
}

// ---------------------------------------------------------------------------
extern "C" void kernel_launch(void* const* d_in, const int* in_sizes, int n_in,
                              void* d_out, int out_size, void* d_ws,
                              size_t ws_size, hipStream_t stream) {
  (void)in_sizes; (void)n_in; (void)out_size; (void)ws_size;
  const float* x  = (const float*)d_in[0];  // [4,2048,4096]  -> [8192,4096]
  const float* W0 = (const float*)d_in[1];  // [4096,4096]
  const float* b0 = (const float*)d_in[2];  // [4096]
  const float* A  = (const float*)d_in[3];  // [4,16,4096]    -> [64,4096]
  const float* Bl = (const float*)d_in[4];  // [4,4096,16]
  float* out = (float*)d_out;

  // workspace layout (~204 MB)
  char* ws = (char*)d_ws;
  size_t off = 0;
  __bf16* xhi = (__bf16*)(ws + off); off += (size_t)MTOT * KTOT * 2;
  __bf16* xlo = (__bf16*)(ws + off); off += (size_t)MTOT * KTOT * 2;
  __bf16* whi = (__bf16*)(ws + off); off += (size_t)NTOT * KTOT * 2;
  __bf16* wlo = (__bf16*)(ws + off); off += (size_t)NTOT * KTOT * 2;
  __bf16* ahi = (__bf16*)(ws + off); off += (size_t)LR * KTOT * 2;
  __bf16* alo = (__bf16*)(ws + off); off += (size_t)LR * KTOT * 2;
  float*  hbf = (float*)(ws + off);  off += (size_t)MTOT * LR * 4;

  olil_split_hilo<<<4096, 256, 0, stream>>>(x, xhi, xlo,
                                            (long long)MTOT * KTOT);
  olil_split_hilo<<<2048, 256, 0, stream>>>(W0, whi, wlo,
                                            (long long)NTOT * KTOT);
  olil_split_hilo<<<64, 256, 0, stream>>>(A, ahi, alo, (long long)LR * KTOT);

  olil_lora_h<<<MTOT / 128, 256, 0, stream>>>(xhi, xlo, ahi, alo, hbf);

  dim3 grid(NTOT / 128, MTOT / 128);
  olil_main_gemm<<<grid, 256, 0, stream>>>(xhi, xlo, whi, wlo, b0, hbf, Bl,
                                           out);
}
